// CfCNCPWrapper_78975858639369
// MI455X (gfx1250) — matrix operand order repaired and checked by reading, not compiled, not run on hardware
//
#include <hip/hip_runtime.h>

typedef _Float16 half_t;
typedef __attribute__((ext_vector_type(16))) _Float16 v16h;
typedef __attribute__((ext_vector_type(8)))  _Float16 v8h;
typedef __attribute__((ext_vector_type(8)))  float    v8f;

// ---------------- geometry ----------------
// layer0: hid=149 cat=277 -> hidp=160 catp=288 (sections x:128 | h0:160)
// layer1: hid= 99 cat=248 -> hidp=112 catp=288 (sections nh0:160 | h1:112 | pad16)
// layer2: hid=  8 cat=107 -> hidp= 16 catp=128 (sections nh1:112 | h2:16)
#define S0W (4*160*288)
#define S1W (4*112*288)
#define S2W (4*16*128)
#define TOTW (S0W + S1W + S2W)      // 321536 halves
#define TOTB (4*160 + 4*112 + 4*16) // 1152 floats
#define YSZ  (256L*512L*8L)

// dynamic LDS layout (halves): [W1|W2 : S1W+S2W][xh0:16*288][xh1:16*288][xh2:16*128]
#define LDSW   (S1W + S2W)                       // 137216 halves
#define LDS_HALVES (LDSW + 16*288 + 16*288 + 16*128)
#define LDS_BYTES  (LDS_HALVES * 2)              // 296960 B  (< 320KB WGP LDS)

struct PrepArgs {
    const float* w[3][4];
    const int*   m[3];
    const float* b[3][4];
};

// ---------------- prep: mask + f16 convert + pad ----------------
__global__ void cfc_prep(PrepArgs pa, half_t* __restrict__ Wout, float* __restrict__ Bout) {
    long idx = (long)blockIdx.x * blockDim.x + threadIdx.x;
    if (idx < TOTW) {
        long r = idx;
        int l, hid, cat, hidp, catp;
        if (r < S0W)            { l = 0; hidp = 160; catp = 288; hid = 149; cat = 277; }
        else if (r < S0W + S1W) { l = 1; r -= S0W; hidp = 112; catp = 288; hid = 99;  cat = 248; }
        else                    { l = 2; r -= S0W + S1W; hidp = 16; catp = 128; hid = 8; cat = 107; }
        int per = hidp * catp;
        int j   = (int)(r / per);
        int rem = (int)(r % per);
        int n   = rem / catp;
        int kp  = rem % catp;
        int col = -1;
        if (l == 0)      { if (kp < 277) col = kp; }
        else if (l == 1) { if (kp < 149) col = kp; else if (kp >= 160 && kp < 259) col = 149 + (kp - 160); }
        else             { if (kp < 99)  col = kp; else if (kp >= 112 && kp < 120) col = 99 + (kp - 112); }
        float v = 0.0f;
        if (n < hid && col >= 0) {
            v = pa.w[l][j][n * cat + col];
            if (j < 2) v *= (float)pa.m[l][n * cat + col];
        }
        Wout[idx] = (half_t)v;
    } else if (idx < TOTW + TOTB) {
        int r = (int)(idx - TOTW);
        int l, hid, hidp;
        if (r < 640)       { l = 0; hidp = 160; hid = 149; }
        else if (r < 1088) { l = 1; r -= 640; hidp = 112; hid = 99; }
        else               { l = 2; r -= 1088; hidp = 16; hid = 8; }
        int j = r / hidp, n = r % hidp;
        Bout[idx - TOTW] = (n < hid) ? pa.b[l][j][n] : 0.0f;
    }
}

// ---------------- WMMA helpers ----------------
__device__ inline v16h load_a_frag(const half_t* xh, int stride, int kb, int m, int hi) {
    const v8h* p = (const v8h*)(xh + m * stride + kb * 32 + hi * 16);
    v8h lo = p[0];
    v8h up = p[1];   // +8 halves
    return __builtin_shufflevector(lo, up, 0,1,2,3,4,5,6,7,8,9,10,11,12,13,14,15);
}

__device__ inline v16h load_b_frag(const half_t* W, int catp, int nt, int kb, int n, int hi) {
    return *(const v16h*)(W + (nt * 16 + n) * catp + kb * 32 + hi * 16);
}

__device__ inline v8f wmma16(v16h a, v16h b, v8f c) {
    return __builtin_amdgcn_wmma_f32_16x16x32_f16(false, a, false, b, (short)0, c, false, false);
}

// W points into global (layer0) or LDS (layers 1/2); address space inferred after inlining.
// `nz` is an opaque zero (loop-carried through asm) that blocks LICM/CSE of the weight loads.
__device__ inline void mm4(const half_t* xh, int stride,
                           const half_t* W, int catp, int hidp,
                           int nt, int ktiles, int n, int hi, int m, int nz,
                           v8f& c1, v8f& c2, v8f& ca, v8f& cb) {
    const int msz = hidp * catp;
    c1 = {}; c2 = {}; ca = {}; cb = {};
    for (int kb = 0; kb < ktiles; ++kb) {
        v16h a  = load_a_frag(xh, stride, kb, m, hi);
        v16h b1 = load_b_frag(W + nz + 0 * msz, catp, nt, kb, n, hi);
        v16h b2 = load_b_frag(W + nz + 1 * msz, catp, nt, kb, n, hi);
        v16h b3 = load_b_frag(W + nz + 2 * msz, catp, nt, kb, n, hi);
        v16h b4 = load_b_frag(W + nz + 3 * msz, catp, nt, kb, n, hi);
        c1 = wmma16(a, b1, c1);
        c2 = wmma16(a, b2, c2);
        ca = wmma16(a, b3, ca);
        cb = wmma16(a, b4, cb);
    }
}

__device__ inline float fast_tanh(float x) { return 1.0f - 2.0f / (__expf(2.0f * x) + 1.0f); }
__device__ inline float fast_sig(float x)  { return 1.0f / (1.0f + __expf(-x)); }

// ---------------- main persistent recurrent kernel ----------------
__global__ __launch_bounds__(320) void cfc_main(
        const float* __restrict__ x, const float* __restrict__ dtp,
        const half_t* __restrict__ W0, const half_t* __restrict__ W1g,
        const float* __restrict__ Bias, float* __restrict__ out) {
    extern __shared__ half_t smem[];
    half_t* Wl  = smem;                 // W1 (4x112x288) then W2 (4x16x128), LDS-resident
    half_t* xh0 = smem + LDSW;          // [x:128 | h0:160]
    half_t* xh1 = xh0 + 16 * 288;       // [nh0:160 | h1:112 | pad16]
    half_t* xh2 = xh1 + 16 * 288;       // [nh1:112 | h2:16]

    const int tid  = threadIdx.x;
    const int wave = tid >> 5;
    const int lane = tid & 31;
    const int n    = lane & 15;
    const int hi   = lane >> 4;
    const int wg   = blockIdx.x;
    const float ts = dtp[0];

    // one-time: stage layer1+layer2 weights into LDS (contiguous in ws)
    for (int i = tid; i < LDSW / 8; i += 320)
        ((v8h*)Wl)[i] = ((const v8h*)W1g)[i];
    // zero activation buffers (initial h = 0, plus all pad lanes)
    for (int i = tid; i < 16 * 288; i += 320) { xh0[i] = (half_t)0; xh1[i] = (half_t)0; }
    for (int i = tid; i < 16 * 128; i += 320) xh2[i] = (half_t)0;
    // stage x(t=0); prefetch x(t=1)
    for (int i = tid; i < 16 * 32; i += 320) {
        int r = i >> 5, c4 = i & 31;
        const float4 v = ((const float4*)(x + ((long)(wg * 16 + r) * 512) * 128))[c4];
        half_t* p = xh0 + r * 288 + c4 * 4;
        p[0] = (half_t)v.x; p[1] = (half_t)v.y; p[2] = (half_t)v.z; p[3] = (half_t)v.w;
    }
    if (tid < 64) {
        int r = tid >> 2, seg = tid & 3;
        __builtin_prefetch(x + ((long)(wg * 16 + r) * 512 + 1) * 128 + seg * 32, 0, 1);
    }

    // per-lane biases, hoisted out of the time loop
    const int col0 = wave * 16 + n;
    const float b01 = Bias[col0],       b02 = Bias[160 + col0];
    const float b03 = Bias[320 + col0], b04 = Bias[480 + col0];
    const int col1 = (wave < 7 ? wave : 0) * 16 + n;
    const float b11 = Bias[640 + col1],       b12 = Bias[640 + 112 + col1];
    const float b13 = Bias[640 + 224 + col1], b14 = Bias[640 + 336 + col1];
    const float b21 = Bias[1088 + n],      b22 = Bias[1088 + 16 + n];
    const float b23 = Bias[1088 + 32 + n], b24 = Bias[1088 + 48 + n];

    __syncthreads();

    int opaque = 0;   // stays 0; asm below makes it loop-carried & unknowable
    for (int t = 0; t < 512; ++t) {
        asm volatile("" : "+s"(opaque));
        const int nz = opaque;

        // ---- phase A: layer0, 10 N-tiles, one per wave; weights stream from L2 ----
        {
            v8f c1, c2, ca, cb;
            mm4(xh0, 288, W0, 288, 160, wave, 9, n, hi, n, nz, c1, c2, ca, cb);
            for (int v = 0; v < 8; ++v) {
                int m = v + 8 * hi;
                float f1 = fast_tanh(c1[v] + b01);
                float f2 = fast_tanh(c2[v] + b02);
                float ti = fast_sig((ca[v] + b03) * ts + (cb[v] + b04));
                float o  = f1 * (1.0f - ti) + ti * f2;
                if (col0 < 149) {
                    xh1[m * 288 + col0] = (half_t)o;
                    if (t == 511) out[YSZ + (long)(wg * 16 + m) * 256 + col0] = o;
                }
            }
        }
        __syncthreads();

        // ---- phase B: waves 0-6 layer1 (LDS weights);
        //      waves 7-9: feed nh0 back into xh0, stage x(t+1), prefetch x(t+2) ----
        if (wave < 7) {
            v8f c1, c2, ca, cb;
            mm4(xh1, 288, Wl, 288, 112, wave, 9, n, hi, n, nz, c1, c2, ca, cb);
            for (int v = 0; v < 8; ++v) {
                int m = v + 8 * hi;
                float f1 = fast_tanh(c1[v] + b11);
                float f2 = fast_tanh(c2[v] + b12);
                float ti = fast_sig((ca[v] + b13) * ts + (cb[v] + b14));
                float o  = f1 * (1.0f - ti) + ti * f2;
                if (col1 < 99) {
                    xh2[m * 128 + col1] = (half_t)o;
                    if (t == 511) out[YSZ + (long)(wg * 16 + m) * 256 + 149 + col1] = o;
                }
            }
        } else {
            int id = (wave - 7) * 32 + lane;                  // 96 lanes
            for (int i = id; i < 320; i += 96) {              // nh0 -> xh0 h-section
                int r = i / 20, ch = i % 20;
                *(v8h*)(xh0 + r * 288 + 128 + ch * 8) = *(const v8h*)(xh1 + r * 288 + ch * 8);
            }
            if (t < 511) {                                    // stage x(t+1), off critical path
                for (int i = id; i < 16 * 32; i += 96) {
                    int r = i >> 5, c4 = i & 31;
                    const float4 v = ((const float4*)(x + ((long)(wg * 16 + r) * 512 + (t + 1)) * 128))[c4];
                    half_t* p = xh0 + r * 288 + c4 * 4;
                    p[0] = (half_t)v.x; p[1] = (half_t)v.y; p[2] = (half_t)v.z; p[3] = (half_t)v.w;
                }
                if (t < 510 && id < 64) {                     // prefetch x(t+2)
                    int r = id >> 2, seg = id & 3;
                    __builtin_prefetch(x + ((long)(wg * 16 + r) * 512 + (t + 2)) * 128 + seg * 32, 0, 1);
                }
            }
        }
        __syncthreads();

        // ---- phase C: wave 0 layer2 (LDS weights); waves 1-3 feed nh1 back into xh1 ----
        if (wave == 0) {
            v8f c1, c2, ca, cb;
            mm4(xh2, 128, Wl + S1W, 128, 16, 0, 4, n, hi, n, nz, c1, c2, ca, cb);
            for (int v = 0; v < 8; ++v) {
                int m = v + 8 * hi;
                float f1 = fast_tanh(c1[v] + b21);
                float f2 = fast_tanh(c2[v] + b22);
                float ti = fast_sig((ca[v] + b23) * ts + (cb[v] + b24));
                float o  = f1 * (1.0f - ti) + ti * f2;
                if (n < 8) {
                    out[((long)(wg * 16 + m) * 512 + t) * 8 + n] = fast_tanh(o);
                    xh2[m * 128 + 112 + n] = (half_t)o;
                    if (t == 511) out[YSZ + (long)(wg * 16 + m) * 256 + 248 + n] = o;
                }
            }
        } else if (wave <= 3) {
            int id = (wave - 1) * 32 + lane;                  // 96 lanes
            for (int i = id; i < 224; i += 96) {              // nh1 -> xh1 h-section
                int r = i / 14, ch = i % 14;
                *(v8h*)(xh1 + r * 288 + 160 + ch * 8) = *(const v8h*)(xh2 + r * 128 + ch * 8);
            }
        }
        __syncthreads();
    }
}

// ---------------- host launcher ----------------
extern "C" void kernel_launch(void* const* d_in, const int* in_sizes, int n_in,
                              void* d_out, int out_size, void* d_ws, size_t ws_size,
                              hipStream_t stream) {
    PrepArgs pa;
    for (int l = 0; l < 3; ++l) {
        for (int j = 0; j < 4; ++j) pa.w[l][j] = (const float*)d_in[l * 9 + j];
        for (int j = 0; j < 4; ++j) pa.b[l][j] = (const float*)d_in[l * 9 + 4 + j];
        pa.m[l] = (const int*)d_in[l * 9 + 8];
    }
    const float* x  = (const float*)d_in[27];
    const float* dt = (const float*)d_in[28];

    half_t* Wout = (half_t*)d_ws;
    float*  Bout = (float*)((char*)d_ws + (size_t)TOTW * sizeof(half_t));

    int total = TOTW + TOTB;
    cfc_prep<<<(total + 255) / 256, 256, 0, stream>>>(pa, Wout, Bout);

    const half_t* W0 = Wout;
    const half_t* W1 = Wout + S0W;      // W1 and W2 contiguous: staged to LDS together
    cfc_main<<<16, 320, LDS_BYTES, stream>>>(x, dt, W0, W1, (const float*)Bout, (float*)d_out);
}
